// EncoderDecoderGNN_31714038513928
// MI455X (gfx1250) — compile-verified, matrix operand order checked by reading, not run on hardware
//
#include <hip/hip_runtime.h>
#include <hip/hip_bf16.h>

// ---------------------------------------------------------------------------
// CDNA5 (gfx1250) implementation of the EncoderDecoderGNN reference.
// All dense layers run as f16 WMMA GEMMs (v_wmma_f32_16x16x32_f16) with fp32
// accumulation; LayerNorm+ReLU fused via an LDS row reduction.
// ---------------------------------------------------------------------------

typedef __attribute__((ext_vector_type(16))) _Float16 v16h;
typedef __attribute__((ext_vector_type(8)))  _Float16 v8h;
typedef __attribute__((ext_vector_type(8)))  float    v8f;

#define N_NODES 32
#define BATCH   1024
#define N_EDGES 128
#define RN      (N_NODES * BATCH)   // 32768 node rows
#define RE      (N_EDGES * BATCH)   // 131072 edge rows
#define FH      256                 // hidden width
#define N_ITERS 6

__device__ __forceinline__ v16h cat16(v8h lo, v8h hi) {
  return __builtin_shufflevector(lo, hi, 0,1,2,3,4,5,6,7,8,9,10,11,12,13,14,15);
}

__device__ __forceinline__ v8f wmma_f16(v16h a, v16h b, v8f c) {
  // D = A(16x32 f16) * B(32x16 f16) + C(16x16 f32)
  return __builtin_amdgcn_wmma_f32_16x16x32_f16(false, a, false, b, (short)0, c,
                                                false, false);
}

// ---------------------------------------------------------------------------
// Weight pre-pack: fp32 W[Kreal x F] -> fp16 packed fragments.
// For each (kstep, column n): 32 halves ordered so lane L (N = L&15,
// group = L>>4) reads its 16 WMMA-B halves as two contiguous v8h loads.
// K offsets per entry idx: grp = idx>>4, h = idx&15,
//   koff = (h<8 ? h : 16 + (h-8)) + 8*grp           (matches ISA 16-bit layout)
// ---------------------------------------------------------------------------
__global__ void __launch_bounds__(256)
pack_w(const float* __restrict__ W, _Float16* __restrict__ Wp,
       int Kreal, int Kpad, int F, int total) {
  int idx = blockIdx.x * 256 + threadIdx.x;
  if (idx >= total) return;
  int within = idx & 31;
  int cn     = idx >> 5;
  int n      = cn % F;
  int ks     = cn / F;
  int grp = within >> 4;
  int h   = within & 15;
  int koff = (h < 8 ? h : 16 + (h - 8)) + 8 * grp;
  int k = ks * 32 + koff;
  float v = (k < Kreal) ? W[(size_t)k * F + n] : 0.f;
  Wp[idx] = (_Float16)v;
}

// ---------------------------------------------------------------------------
// node_input: [N,B,13] padded to fp16 [RN,32]
// cols 0-4 = state[b,0:5]; 5 = state[b,5+n]; 6 = state[b,5+N+n]; 7-12 = feat
// ---------------------------------------------------------------------------
__global__ void __launch_bounds__(256)
build_node_in(const float* __restrict__ state, const float* __restrict__ nodef,
              _Float16* __restrict__ Xn) {
  int idx = blockIdx.x * 256 + threadIdx.x;   // RN*32 threads
  int r = idx >> 5, c = idx & 31;
  int n = r >> 10, bb = r & 1023;
  const int SR = 5 + 2 * N_NODES;
  float v = 0.f;
  if (c < 5)       v = state[bb * SR + c];
  else if (c == 5) v = state[bb * SR + 5 + n];
  else if (c == 6) v = state[bb * SR + 5 + N_NODES + n];
  else if (c < 13) v = nodef[n * 6 + (c - 7)];
  Xn[idx] = (_Float16)v;
}

// per-node incoming edge lists from edge_dst
__global__ void build_elist(const int* __restrict__ edge_dst,
                            int* __restrict__ elist, int* __restrict__ ecount) {
  int n = threadIdx.x;
  if (n >= N_NODES) return;
  int cnt = 0;
  for (int e = 0; e < N_EDGES; ++e)
    if (edge_dst[e] == n) elist[n * N_EDGES + cnt++] = e;
  ecount[n] = cnt;
}

// msg_in fp16 [RE,96]: [h[src] (64) | edge_feat (1) | node_input[src] (13) | 0 pad]
__global__ void __launch_bounds__(256)
build_msg(const _Float16* __restrict__ h, const _Float16* __restrict__ Xn,
          const float* __restrict__ efeat, const int* __restrict__ esrc,
          _Float16* __restrict__ M) {
  int idx = blockIdx.x * 256 + threadIdx.x;
  if (idx >= RE * 96) return;
  int r = idx / 96, c = idx - r * 96;
  int e = r >> 10, bb = r & 1023;
  int src = esrc[e];
  _Float16 v = (_Float16)0.f;
  if (c < 64)       v = h[((size_t)(src * BATCH + bb)) * 64 + c];
  else if (c == 64) v = (_Float16)efeat[e];
  else if (c < 78)  v = Xn[((size_t)(src * BATCH + bb)) * 32 + (c - 65)];
  M[idx] = v;
}

// segment_max over incoming edges + concat h -> U[RN,128]
__global__ void __launch_bounds__(256)
segmax_concat(const _Float16* __restrict__ m, const _Float16* __restrict__ h,
              const int* __restrict__ elist, const int* __restrict__ ecount,
              _Float16* __restrict__ U) {
  int idx = blockIdx.x * 256 + threadIdx.x;   // RN*128 threads
  int r = idx >> 7, c = idx & 127;
  int n = r >> 10, bb = r & 1023;
  if (c < 64) {
    int cnt = ecount[n];
    float best = -3.4e38f;
    for (int i = 0; i < cnt; ++i) {
      int e = elist[n * N_EDGES + i];
      float v = (float)m[((size_t)(e * BATCH + bb)) * 64 + c];
      best = fmaxf(best, v);
    }
    U[(size_t)r * 128 + c] = (_Float16)best;
  } else {
    U[(size_t)r * 128 + c] = h[(size_t)r * 64 + (c - 64)];
  }
}

// ---------------------------------------------------------------------------
// Fused GEMM + bias + LayerNorm + ReLU.  Y[rows,256] = relu(LN(X@W + b))
// Block = 256 threads (8 waves) owns a 16x256 output tile.
// Wave w computes column tiles [32w,32w+16) and [32w+16,32w+32).
// ---------------------------------------------------------------------------
template<int KPAD>
__global__ void __launch_bounds__(256)
gemm_ln_relu(const _Float16* __restrict__ X, const _Float16* __restrict__ Wp,
             const float* __restrict__ bias, const float* __restrict__ gamma,
             const float* __restrict__ beta, _Float16* __restrict__ Y) {
  __shared__ float tile[16][FH + 4];
  __shared__ float psum[256], psq[256];
  __shared__ float mu_s[16], rs_s[16];

  const int t    = threadIdx.x;
  const int lane = t & 31;
  const int wave = t >> 5;
  const int hi   = lane >> 4;      // lane group (selects K sub-range)
  const int l16  = lane & 15;      // M for A, N for B/C/D
  const size_t rowTile = (size_t)blockIdx.x * 16;

  const _Float16* xp = X + (rowTile + l16) * KPAD + hi * 8;
  const int n0 = wave * 32, n1 = n0 + 16;
  v8f acc0 = {0.f,0.f,0.f,0.f,0.f,0.f,0.f,0.f};
  v8f acc1 = acc0;

#pragma unroll
  for (int ks = 0; ks < KPAD / 32; ++ks) {
    v16h a = cat16(*(const v8h*)(xp + ks * 32), *(const v8h*)(xp + ks * 32 + 16));
    const _Float16* pb0 = Wp + ((size_t)(ks * FH + n0 + l16) << 5) + (hi << 4);
    const _Float16* pb1 = Wp + ((size_t)(ks * FH + n1 + l16) << 5) + (hi << 4);
    v16h b0 = cat16(*(const v8h*)pb0, *(const v8h*)(pb0 + 8));
    v16h b1 = cat16(*(const v8h*)pb1, *(const v8h*)(pb1 + 8));
    acc0 = wmma_f16(a, b0, acc0);
    acc1 = wmma_f16(a, b1, acc1);
  }

  // scatter accumulators (+bias) to LDS:  reg r -> row = r + 8*hi
#pragma unroll
  for (int r = 0; r < 8; ++r) {
    int row = r + hi * 8;
    tile[row][n0 + l16] = acc0[r] + bias[n0 + l16];
    tile[row][n1 + l16] = acc1[r] + bias[n1 + l16];
  }
  __syncthreads();

  // LayerNorm over 256 features per row: two-level reduction
  const int prow = t >> 4;
  const int pc0  = (t & 15) << 4;
  float s = 0.f, s2 = 0.f;
#pragma unroll
  for (int j = 0; j < 16; ++j) { float v = tile[prow][pc0 + j]; s += v; s2 += v * v; }
  psum[t] = s; psq[t] = s2;
  __syncthreads();
  if (t < 16) {
    float S = 0.f, S2 = 0.f;
#pragma unroll
    for (int j = 0; j < 16; ++j) { S += psum[(t << 4) + j]; S2 += psq[(t << 4) + j]; }
    float mu  = S * (1.f / FH);
    float var = S2 * (1.f / FH) - mu * mu;
    mu_s[t] = mu;
    rs_s[t] = rsqrtf(var + 1e-5f);
  }
  __syncthreads();

  const float mu = mu_s[prow], rs = rs_s[prow];
  _Float16* yp = Y + (rowTile + prow) * FH + pc0;
#pragma unroll
  for (int j = 0; j < 16; ++j) {
    int c = pc0 + j;
    float v = (tile[prow][c] - mu) * rs * gamma[c] + beta[c];
    yp[j] = (_Float16)fmaxf(v, 0.f);
  }
}

// ---------------------------------------------------------------------------
// Final projection: Y[rows,64] = act(X[rows,256] @ W[256,64] + b)
// Block = 8 waves covers 32 rows x 64 cols; one 16x16 WMMA tile per wave.
// ACT: 0 = identity, 1 = tanh
// ---------------------------------------------------------------------------
template<int ACT>
__global__ void __launch_bounds__(256)
gemm_out64(const _Float16* __restrict__ X, const _Float16* __restrict__ Wp,
           const float* __restrict__ bias, _Float16* __restrict__ Y) {
  const int t    = threadIdx.x;
  const int lane = t & 31;
  const int wave = t >> 5;
  const int hi   = lane >> 4;
  const int l16  = lane & 15;
  const size_t rowTile = (size_t)blockIdx.x * 32 + (size_t)(wave >> 2) * 16;
  const int n0 = (wave & 3) * 16;

  const _Float16* xp = X + (rowTile + l16) * FH + hi * 8;
  v8f acc = {0.f,0.f,0.f,0.f,0.f,0.f,0.f,0.f};
#pragma unroll
  for (int ks = 0; ks < FH / 32; ++ks) {
    v16h a = cat16(*(const v8h*)(xp + ks * 32), *(const v8h*)(xp + ks * 32 + 16));
    const _Float16* pb = Wp + ((size_t)(ks * 64 + n0 + l16) << 5) + (hi << 4);
    v16h b = cat16(*(const v8h*)pb, *(const v8h*)(pb + 8));
    acc = wmma_f16(a, b, acc);
  }
  const int col = n0 + l16;
  const float bv = bias[col];
#pragma unroll
  for (int r = 0; r < 8; ++r) {
    size_t row = rowTile + r + hi * 8;
    float v = acc[r] + bv;
    if (ACT == 1) v = tanhf(v);
    Y[row * 64 + col] = (_Float16)v;
  }
}

// decoder head: out = normalize([tanh(h@Wo+bo), sigmoid(h@Ws+bs)])
__global__ void __launch_bounds__(256)
final_out(const _Float16* __restrict__ h2, const float* __restrict__ wo,
          const float* __restrict__ ws, const float* __restrict__ bo,
          const float* __restrict__ bs, float* __restrict__ out) {
  int r = blockIdx.x * 256 + threadIdx.x;   // RN threads
  const _Float16* hp = h2 + (size_t)r * FH;
  float s0 = 0.f, s1 = 0.f;
  for (int k = 0; k < FH; ++k) {
    float x = (float)hp[k];
    s0 += x * wo[k];
    s1 += x * ws[k];
  }
  float o0 = tanhf(s0 + bo[0]);
  float o1 = 1.f / (1.f + expf(-(s1 + bs[0])));
  float nrm = fmaxf(sqrtf(o0 * o0 + o1 * o1), 1e-12f);
  out[(size_t)r * 2]     = o0 / nrm;
  out[(size_t)r * 2 + 1] = o1 / nrm;
}

// ---------------------------------------------------------------------------
extern "C" void kernel_launch(void* const* d_in, const int* in_sizes, int n_in,
                              void* d_out, int out_size, void* d_ws, size_t ws_size,
                              hipStream_t stream) {
  (void)in_sizes; (void)n_in; (void)out_size; (void)ws_size;
  // jax pytree flatten: dict keys alphabetical -> input, message, output, update
  // subnet keys: Wo,(Ws),bo,(bs),hidden[ W,b,beta,g ]
  auto F32 = [&](int i) { return (const float*)d_in[i]; };
  const float* W_in_o = F32(0);   const float* b_in_o = F32(1);
  const float* W_in_0 = F32(2);   const float* b_in_0 = F32(3);
  const float* be_in_0 = F32(4);  const float* g_in_0 = F32(5);
  const float* W_in_1 = F32(6);   const float* b_in_1 = F32(7);
  const float* be_in_1 = F32(8);  const float* g_in_1 = F32(9);
  const float* W_ms_o = F32(10);  const float* b_ms_o = F32(11);
  const float* W_ms_0 = F32(12);  const float* b_ms_0 = F32(13);
  const float* be_ms_0 = F32(14); const float* g_ms_0 = F32(15);
  const float* W_ms_1 = F32(16);  const float* b_ms_1 = F32(17);
  const float* be_ms_1 = F32(18); const float* g_ms_1 = F32(19);
  const float* W_ou_o = F32(20);  const float* W_ou_s = F32(21);
  const float* b_ou_o = F32(22);  const float* b_ou_s = F32(23);
  const float* W_ou_0 = F32(24);  const float* b_ou_0 = F32(25);
  const float* be_ou_0 = F32(26); const float* g_ou_0 = F32(27);
  const float* W_ou_1 = F32(28);  const float* b_ou_1 = F32(29);
  const float* be_ou_1 = F32(30); const float* g_ou_1 = F32(31);
  const float* W_up_o = F32(32);  const float* b_up_o = F32(33);
  const float* W_up_0 = F32(34);  const float* b_up_0 = F32(35);
  const float* be_up_0 = F32(36); const float* g_up_0 = F32(37);
  const float* W_up_1 = F32(38);  const float* b_up_1 = F32(39);
  const float* be_up_1 = F32(40); const float* g_up_1 = F32(41);
  const float* state = F32(42);
  const float* nodef = F32(43);
  const float* efeat = F32(44);
  const int*   esrc  = (const int*)d_in[45];
  const int*   edst  = (const int*)d_in[46];

  // deterministic workspace carve-out (256B aligned)
  char* p = (char*)d_ws;
  auto carve  = [&](size_t bytes) -> char* {
    char* r = p; p += (bytes + 255) & ~(size_t)255; return r;
  };
  auto carveH = [&](size_t nh) -> _Float16* {
    return (_Float16*)carve(nh * sizeof(_Float16));
  };
  _Float16* Wp_in0 = carveH(32  * 256);
  _Float16* Wp_in1 = carveH(256 * 256);
  _Float16* Wp_ino = carveH(256 * 64);
  _Float16* Wp_ms0 = carveH(96  * 256);
  _Float16* Wp_ms1 = carveH(256 * 256);
  _Float16* Wp_mso = carveH(256 * 64);
  _Float16* Wp_up0 = carveH(128 * 256);
  _Float16* Wp_up1 = carveH(256 * 256);
  _Float16* Wp_upo = carveH(256 * 64);
  _Float16* Wp_ou0 = carveH(64  * 256);
  _Float16* Wp_ou1 = carveH(256 * 256);
  _Float16* Xn  = carveH((size_t)RN * 32);
  _Float16* H   = carveH((size_t)RN * 64);
  _Float16* Min = carveH((size_t)RE * 96);
  _Float16* A1  = carveH((size_t)RE * 256);
  _Float16* A2  = carveH((size_t)RE * 256);
  _Float16* Mv  = carveH((size_t)RE * 64);
  _Float16* Uin = carveH((size_t)RN * 128);
  int* elist  = (int*)carve((size_t)N_NODES * N_EDGES * sizeof(int));
  int* ecount = (int*)carve((size_t)N_NODES * sizeof(int));

  auto pack = [&](const float* W, _Float16* Wp, int Kreal, int Kpad, int Fo) {
    int total = Kpad * Fo;
    pack_w<<<(total + 255) / 256, 256, 0, stream>>>(W, Wp, Kreal, Kpad, Fo, total);
  };
  pack(W_in_0, Wp_in0, 13,  32,  256);
  pack(W_in_1, Wp_in1, 256, 256, 256);
  pack(W_in_o, Wp_ino, 256, 256, 64);
  pack(W_ms_0, Wp_ms0, 78,  96,  256);
  pack(W_ms_1, Wp_ms1, 256, 256, 256);
  pack(W_ms_o, Wp_mso, 256, 256, 64);
  pack(W_up_0, Wp_up0, 128, 128, 256);
  pack(W_up_1, Wp_up1, 256, 256, 256);
  pack(W_up_o, Wp_upo, 256, 256, 64);
  pack(W_ou_0, Wp_ou0, 64,  64,  256);
  pack(W_ou_1, Wp_ou1, 256, 256, 256);

  build_node_in<<<RN * 32 / 256, 256, 0, stream>>>(state, nodef, Xn);
  build_elist<<<1, 32, 0, stream>>>(edst, elist, ecount);

  // encoder: node_input -> h
  gemm_ln_relu<32 ><<<RN / 16, 256, 0, stream>>>(Xn, Wp_in0, b_in_0, g_in_0, be_in_0, A1);
  gemm_ln_relu<256><<<RN / 16, 256, 0, stream>>>(A1, Wp_in1, b_in_1, g_in_1, be_in_1, A2);
  gemm_out64<0><<<RN / 32, 256, 0, stream>>>(A2, Wp_ino, b_in_o, H);

  // message passing iterations
  for (int it = 0; it < N_ITERS; ++it) {
    build_msg<<<(RE * 96 + 255) / 256, 256, 0, stream>>>(H, Xn, efeat, esrc, Min);
    gemm_ln_relu<96 ><<<RE / 16, 256, 0, stream>>>(Min, Wp_ms0, b_ms_0, g_ms_0, be_ms_0, A1);
    gemm_ln_relu<256><<<RE / 16, 256, 0, stream>>>(A1, Wp_ms1, b_ms_1, g_ms_1, be_ms_1, A2);
    gemm_out64<1><<<RE / 32, 256, 0, stream>>>(A2, Wp_mso, b_ms_o, Mv);       // tanh
    segmax_concat<<<RN * 128 / 256, 256, 0, stream>>>(Mv, H, elist, ecount, Uin);
    gemm_ln_relu<128><<<RN / 16, 256, 0, stream>>>(Uin, Wp_up0, b_up_0, g_up_0, be_up_0, A1);
    gemm_ln_relu<256><<<RN / 16, 256, 0, stream>>>(A1, Wp_up1, b_up_1, g_up_1, be_up_1, A2);
    gemm_out64<0><<<RN / 32, 256, 0, stream>>>(A2, Wp_upo, b_up_o, H);
  }

  // decoder
  gemm_ln_relu<64 ><<<RN / 16, 256, 0, stream>>>(H,  Wp_ou0, b_ou_0, g_ou_0, be_ou_0, A1);
  gemm_ln_relu<256><<<RN / 16, 256, 0, stream>>>(A1, Wp_ou1, b_ou_1, g_ou_1, be_ou_1, A2);
  final_out<<<RN / 256, 256, 0, stream>>>(A2, W_ou_o, W_ou_s, b_ou_o, b_ou_s,
                                          (float*)d_out);
}
